// AttentionCNN_31344671326349
// MI455X (gfx1250) — matrix operand order repaired and checked by reading, not compiled, not run on hardware
//
#include <hip/hip_runtime.h>
#include <stddef.h>

// ---------------------------------------------------------------------------
// AttentionCNN pipeline for MI455X (gfx1250), wave32 + WMMA bf16.
// B=256 S=3000 C=64 E=512 A=256 NC=2
// ws layout: enc bf16 16MiB | q bf16 8MiB | k bf16 8MiB | v bf16 8MiB | att f32 16MiB
// ---------------------------------------------------------------------------

typedef __attribute__((ext_vector_type(16))) __bf16       v16bf;
typedef __attribute__((ext_vector_type(8)))  float        v8f;
typedef __attribute__((ext_vector_type(4)))  unsigned int u32x4;
typedef __attribute__((ext_vector_type(4)))  float        f32x4;
typedef __attribute__((ext_vector_type(8)))  __bf16       bf16x8;
typedef int i32x4v __attribute__((vector_size(16)));   // matches builtin proto

#define BDIM 256
#define SDIM 3000
#define CDIM 64
#define EDIM 512
#define ADIM 256

#ifndef __has_builtin
#define __has_builtin(x) 0
#endif
#if defined(__HIP_DEVICE_COMPILE__) && \
    __has_builtin(__builtin_amdgcn_global_load_async_to_lds_b128) && \
    __has_builtin(__builtin_amdgcn_s_wait_asynccnt)
#define HAVE_ASYNC_LDS 1
#else
#define HAVE_ASYNC_LDS 0
#endif

#define AS1 __attribute__((address_space(1)))
#define AS3 __attribute__((address_space(3)))

// 16-byte global -> LDS copy; async (ASYNCcnt) when the gfx1250 builtin exists.
__device__ __forceinline__ void copy16_g2l(const void* gsrc, void* ldst) {
#if HAVE_ASYNC_LDS
    __builtin_amdgcn_global_load_async_to_lds_b128(
        (AS1 i32x4v*)gsrc, (AS3 i32x4v*)ldst, 0, 0);
#else
    *(u32x4*)ldst = *(const u32x4*)gsrc;
#endif
}
__device__ __forceinline__ void g2l_wait_all() {
#if HAVE_ASYNC_LDS
    __builtin_amdgcn_s_wait_asynccnt(0);
#endif
}

// Native f32 -> bf16 (RNE); lowers to gfx1250 hardware convert instead of
// the 3-VALU-op manual rounding sequence.
__device__ __forceinline__ __bf16 f2bf(float f) {
    return (__bf16)f;
}

__device__ __forceinline__ v8f wmma_bf16(v16bf a, v16bf b, v8f c) {
    return __builtin_amdgcn_wmma_f32_16x16x32_bf16(
        false, a, false, b, (short)0, c, false, false);
}

// A fragment (16x32, M x K) from row-major [row][0..31], stride in halfs.
// Lane l<16: row=l, K {0..7, 16..23}. Lane l>=16: row=l-16, K {8..15, 24..31}.
__device__ __forceinline__ v16bf load_frag_a(const __bf16* base, int stride,
                                             int row, int lane) {
    const __bf16* p = base + row * stride + ((lane & 16) ? 8 : 0);
    union { v16bf v; u32x4 u[2]; } f;
    f.u[0] = *(const u32x4*)(p);
    f.u[1] = *(const u32x4*)(p + 16);
    return f.v;
}

// B fragment (32x16, K x N) from N-major storage [n][0..31] (B^T row-major ==
// WMMA B column-major). Lane l<16: col=l, K {0..7, 16..23}. Lane l>=16: K {8..15, 24..31}.
__device__ __forceinline__ v16bf load_frag_b(const __bf16* base, int stride,
                                             int row, int lane) {
    const __bf16* p = base + row * stride + ((lane & 16) ? 8 : 0);
    union { v16bf v; u32x4 u[2]; } f;
    f.u[0] = *(const u32x4*)(p);
    f.u[1] = *(const u32x4*)(p + 16);
    return f.v;
}

// ---------------------------------------------------------------------------
// Kernel 1: encoder.  For channel c: enc[b,c,:] = relu(x[b,:,c] @ W[c].T + b)
// Per-channel GEMM M=256(b) K=3000(s) N=512(e). Block tile 128x128, BK=32,
// double-buffered LDS (1 barrier per K-step, staging overlaps WMMA).
// ---------------------------------------------------------------------------
__global__ __launch_bounds__(256) void enc_kernel(
    const float* __restrict__ x,      // (B,S,C)
    const float* __restrict__ W,      // (C,E,S)
    const float* __restrict__ bias,   // (C,E)
    __bf16* __restrict__ enc)         // (B*C, E)
{
    __shared__ __align__(16) __bf16 sA[2][128 * 40];
    __shared__ __align__(16) __bf16 sB[2][128 * 40];

    const int m0 = blockIdx.x * 128;          // b offset
    const int n0 = blockIdx.y * 128;          // e offset
    const int c  = blockIdx.z;
    const int tid  = threadIdx.x;
    const int lane = tid & 31;
    const int w    = tid >> 5;
    const int wm = (w >> 2) * 64;
    const int wn = (w & 3) * 32;

    const int rowS = tid >> 1;                // 0..127 staging row
    const int kb   = (tid & 1) * 16;          // 0 or 16

    const float* xrow = x + ((size_t)(m0 + rowS) * SDIM) * CDIM + c;
    const float* wrow = W + ((size_t)c * EDIM + (n0 + rowS)) * SDIM;

    auto stage = [&](int k0, __bf16* aBuf, __bf16* bBuf) {
        // A: x[b, s, c] strided by C, f32 -> bf16, zero-pad K tail
        #pragma unroll
        for (int j = 0; j < 16; ++j) {
            int s = k0 + kb + j;
            float val = (s < SDIM) ? xrow[(size_t)s * CDIM] : 0.0f;
            aBuf[rowS * 40 + kb + j] = f2bf(val);
        }
        // B: W[c, e, s] contiguous in s, vectorized f32 -> bf16
        if (k0 + 64 < SDIM)                    // prefetch 2 tiles ahead
            __builtin_prefetch(wrow + k0 + 64 + kb, 0, 3);
        #pragma unroll
        for (int j = 0; j < 16; j += 4) {
            int s = k0 + kb + j;
            f32x4 f;
            if (s + 3 < SDIM) {
                f = *(const f32x4*)(wrow + s);
            } else {
                f[0] = (s + 0 < SDIM) ? wrow[s + 0] : 0.0f;
                f[1] = (s + 1 < SDIM) ? wrow[s + 1] : 0.0f;
                f[2] = (s + 2 < SDIM) ? wrow[s + 2] : 0.0f;
                f[3] = (s + 3 < SDIM) ? wrow[s + 3] : 0.0f;
            }
            __bf16* d = &bBuf[rowS * 40 + kb + j];
            d[0] = f2bf(f[0]); d[1] = f2bf(f[1]);
            d[2] = f2bf(f[2]); d[3] = f2bf(f[3]);
        }
    };

    v8f acc[4][2] = {};
    const int NT = (SDIM + 31) / 32;          // 94

    stage(0, sA[0], sB[0]);
    for (int kt = 0; kt < NT; ++kt) {
        const int cur = kt & 1;
        __syncthreads();                      // tile kt fully staged
        if (kt + 1 < NT)
            stage((kt + 1) * 32, sA[cur ^ 1], sB[cur ^ 1]);

        v16bf bfr[2];
        #pragma unroll
        for (int nt = 0; nt < 2; ++nt)
            bfr[nt] = load_frag_b(sB[cur], 40, wn + nt * 16 + (lane & 15), lane);
        #pragma unroll
        for (int mt = 0; mt < 4; ++mt) {
            v16bf afr = load_frag_a(sA[cur], 40, wm + mt * 16 + (lane & 15), lane);
            #pragma unroll
            for (int nt = 0; nt < 2; ++nt)
                acc[mt][nt] = wmma_bf16(afr, bfr[nt], acc[mt][nt]);
        }
    }

    // epilogue: + bias, ReLU, bf16 store to enc[(b*C + c)*E + e]
    #pragma unroll
    for (int mt = 0; mt < 4; ++mt) {
        #pragma unroll
        for (int nt = 0; nt < 2; ++nt) {
            int n  = n0 + wn + nt * 16 + (lane & 15);
            int mb = m0 + wm + mt * 16 + ((lane >> 4) << 3);
            float bv = bias[c * EDIM + n];
            #pragma unroll
            for (int r = 0; r < 8; ++r) {
                float val = fmaxf(acc[mt][nt][r] + bv, 0.0f);
                enc[((size_t)(mb + r) * CDIM + c) * EDIM + n] = f2bf(val);
            }
        }
    }
}

// ---------------------------------------------------------------------------
// Kernel 2: q/k/v projection.  out[m, a] = enc[m, :] @ W[a, :] + bias[a]
// M = B*C = 16384, K = E = 512, N = A = 256.  Double-buffered; A-operand
// staged with async global->LDS (pure bf16 copy, ASYNCcnt-tracked).
// ---------------------------------------------------------------------------
__global__ __launch_bounds__(256) void qkv_kernel(
    const __bf16* __restrict__ enc,   // (B*C, E) bf16
    const float*  __restrict__ W,     // (A, E) f32
    const float*  __restrict__ bias,  // (A,)
    __bf16* __restrict__ out)         // (B*C, A) bf16
{
    __shared__ __align__(16) __bf16 sA[2][128 * 40];
    __shared__ __align__(16) __bf16 sB[2][128 * 40];

    const int m0 = blockIdx.x * 128;
    const int n0 = blockIdx.y * 128;
    const int tid  = threadIdx.x;
    const int lane = tid & 31;
    const int w    = tid >> 5;
    const int wm = (w >> 2) * 64;
    const int wn = (w & 3) * 32;

    const int rowS = tid >> 1;
    const int kb   = (tid & 1) * 16;

    auto stage = [&](int k0, __bf16* aBuf, __bf16* bBuf) {
        // A: bf16 copy (async path uses ASYNCcnt-tracked LDS-direct loads)
        const __bf16* src = enc + (size_t)(m0 + rowS) * EDIM + k0 + kb;
        __bf16* dst = aBuf + rowS * 40 + kb;
        copy16_g2l(src,     dst);
        copy16_g2l(src + 8, dst + 8);
        // B: W f32 -> bf16
        const float* wp = W + (size_t)(n0 + rowS) * EDIM + k0 + kb;
        if (k0 + 64 < EDIM)
            __builtin_prefetch(wp + 64, 0, 3);
        __bf16* d = &bBuf[rowS * 40 + kb];
        #pragma unroll
        for (int j = 0; j < 16; j += 4) {
            f32x4 f = *(const f32x4*)(wp + j);
            d[j + 0] = f2bf(f[0]); d[j + 1] = f2bf(f[1]);
            d[j + 2] = f2bf(f[2]); d[j + 3] = f2bf(f[3]);
        }
    };

    v8f acc[4][2] = {};
    const int NT = EDIM / 32;                 // 16

    stage(0, sA[0], sB[0]);
    for (int kt = 0; kt < NT; ++kt) {
        const int cur = kt & 1;
        g2l_wait_all();                       // async A-tile landed in LDS
        __syncthreads();
        if (kt + 1 < NT)
            stage((kt + 1) * 32, sA[cur ^ 1], sB[cur ^ 1]);

        v16bf bfr[2];
        #pragma unroll
        for (int nt = 0; nt < 2; ++nt)
            bfr[nt] = load_frag_b(sB[cur], 40, wn + nt * 16 + (lane & 15), lane);
        #pragma unroll
        for (int mt = 0; mt < 4; ++mt) {
            v16bf afr = load_frag_a(sA[cur], 40, wm + mt * 16 + (lane & 15), lane);
            #pragma unroll
            for (int nt = 0; nt < 2; ++nt)
                acc[mt][nt] = wmma_bf16(afr, bfr[nt], acc[mt][nt]);
        }
    }

    #pragma unroll
    for (int mt = 0; mt < 4; ++mt) {
        #pragma unroll
        for (int nt = 0; nt < 2; ++nt) {
            int n  = n0 + wn + nt * 16 + (lane & 15);
            int mb = m0 + wm + mt * 16 + ((lane >> 4) << 3);
            float bv = bias[n];
            #pragma unroll
            for (int r = 0; r < 8; ++r)
                out[(size_t)(mb + r) * ADIM + n] = f2bf(acc[mt][nt][r] + bv);
        }
    }
}

// ---------------------------------------------------------------------------
// Kernel 3: attention, one workgroup per batch.
// scores = q k^T / 16 ; softmax ; att = probs v   (C=64 tokens, A=256)
// ---------------------------------------------------------------------------
#define QK_STRIDE 264   // 256 + 8 halfs padding (stride = 528 B, 16B multiple)
#define S_STRIDE  66
#define P_STRIDE  72
#define VT_STRIDE 72

__global__ __launch_bounds__(256) void attn_kernel(
    const __bf16* __restrict__ q,
    const __bf16* __restrict__ k,
    const __bf16* __restrict__ v,
    float* __restrict__ att)          // (B*C, A) f32
{
    extern __shared__ __align__(16) char smem[];
    __bf16* sQ  = (__bf16*)smem;                      // 64 x 264
    __bf16* sK  = sQ + 64 * QK_STRIDE;                // 64 x 264
    float*  sS  = (float*)(sK + 64 * QK_STRIDE);      // 64 x 66
    __bf16* sP  = (__bf16*)(sS + 64 * S_STRIDE);      // 64 x 72
    __bf16* sVT = sP + 64 * P_STRIDE;                 // 256 x 72 (v transposed)

    const int b   = blockIdx.x;
    const int tid = threadIdx.x;
    const int lane = tid & 31;
    const int w    = tid >> 5;

    // ---- stage q, k (async 16B copies) and v transposed ----
    {
        int c0 = tid >> 2;                 // 0..63
        int cb = (tid & 3) * 64;           // 0,64,128,192
        const __bf16* srcq = q + ((size_t)b * CDIM + c0) * ADIM + cb;
        const __bf16* srck = k + ((size_t)b * CDIM + c0) * ADIM + cb;
        __bf16* dq = sQ + c0 * QK_STRIDE + cb;
        __bf16* dk = sK + c0 * QK_STRIDE + cb;
        #pragma unroll
        for (int j = 0; j < 8; ++j) {
            copy16_g2l(srcq + j * 8, dq + j * 8);
            copy16_g2l(srck + j * 8, dk + j * 8);
        }

        const bf16x8* vv = (const bf16x8*)(v + ((size_t)b * CDIM + c0) * ADIM + cb);
        #pragma unroll
        for (int jj = 0; jj < 8; ++jj) {
            bf16x8 t8 = vv[jj];
            #pragma unroll
            for (int e = 0; e < 8; ++e)
                sVT[(cb + jj * 8 + e) * VT_STRIDE + c0] = t8[e];
        }
    }
    g2l_wait_all();
    __syncthreads();

    // ---- scores: 4x4 grid of 16x16 tiles, K=256, 2 tiles per wave ----
    #pragma unroll
    for (int i = 0; i < 2; ++i) {
        int t  = w * 2 + i;
        int mt = t >> 2, nt = t & 3;
        v8f acc = {};
        #pragma unroll
        for (int k0 = 0; k0 < ADIM; k0 += 32) {
            v16bf afr = load_frag_a(sQ + k0, QK_STRIDE, mt * 16 + (lane & 15), lane);
            v16bf bfr = load_frag_b(sK + k0, QK_STRIDE, nt * 16 + (lane & 15), lane);
            acc = wmma_bf16(afr, bfr, acc);
        }
        int n  = nt * 16 + (lane & 15);
        int mb = mt * 16 + ((lane >> 4) << 3);
        #pragma unroll
        for (int r = 0; r < 8; ++r)
            sS[(mb + r) * S_STRIDE + n] = acc[r] * 0.0625f;   // 1/sqrt(256)
    }
    __syncthreads();

    // ---- softmax over rows (one thread per row) ----
    if (tid < 64) {
        float mx = -3.0e38f;
        for (int d = 0; d < 64; ++d)
            mx = fmaxf(mx, sS[tid * S_STRIDE + d]);
        float sum = 0.0f;
        for (int d = 0; d < 64; ++d) {
            float e = __expf(sS[tid * S_STRIDE + d] - mx);
            sS[tid * S_STRIDE + d] = e;
            sum += e;
        }
        float inv = 1.0f / sum;
        for (int d = 0; d < 64; ++d)
            sP[tid * P_STRIDE + d] = f2bf(sS[tid * S_STRIDE + d] * inv);
    }
    __syncthreads();

    // ---- att = probs @ v : M=64 K=64 N=256; each wave 16(M) x 128(N) ----
    {
        int mt = w >> 1;
        int nb = (w & 1) * 128;
        v8f acc[8] = {};
        #pragma unroll
        for (int k0 = 0; k0 < 64; k0 += 32) {
            v16bf afr = load_frag_a(sP + k0, P_STRIDE, mt * 16 + (lane & 15), lane);
            #pragma unroll
            for (int nt = 0; nt < 8; ++nt) {
                v16bf bfr = load_frag_b(sVT + k0, VT_STRIDE,
                                        nb + nt * 16 + (lane & 15), lane);
                acc[nt] = wmma_bf16(afr, bfr, acc[nt]);
            }
        }
        float* outp = att + (size_t)b * CDIM * ADIM;
        int mb = mt * 16 + ((lane >> 4) << 3);
        #pragma unroll
        for (int nt = 0; nt < 8; ++nt) {
            int n = nb + nt * 16 + (lane & 15);
            #pragma unroll
            for (int r = 0; r < 8; ++r)
                outp[(size_t)(mb + r) * ADIM + n] = acc[nt][r];
        }
    }
}

// ---------------------------------------------------------------------------
// Kernel 4: conv1d(C->1,k=1) + ReLU + FC(A->NC). One block per batch.
// ---------------------------------------------------------------------------
__global__ __launch_bounds__(256) void conv_fc_kernel(
    const float* __restrict__ att,     // (B*C, A)
    const float* __restrict__ conv_w,  // (C,)
    const float* __restrict__ conv_b,  // (1,)
    const float* __restrict__ fc_w,    // (NC, A)
    const float* __restrict__ fc_b,    // (NC,)
    float* __restrict__ out)           // (B, NC)
{
    __shared__ float red0[256];
    __shared__ float red1[256];
    const int b = blockIdx.x;
    const int a = threadIdx.x;

    const float* ap = att + (size_t)b * CDIM * ADIM + a;
    float s = 0.0f;
    #pragma unroll 8
    for (int c = 0; c < CDIM; ++c)
        s += ap[(size_t)c * ADIM] * conv_w[c];
    s = fmaxf(s + conv_b[0], 0.0f);

    red0[a] = s * fc_w[a];
    red1[a] = s * fc_w[ADIM + a];
    __syncthreads();
    for (int off = 128; off > 0; off >>= 1) {
        if (a < off) { red0[a] += red0[a + off]; red1[a] += red1[a + off]; }
        __syncthreads();
    }
    if (a == 0) {
        out[b * 2 + 0] = red0[0] + fc_b[0];
        out[b * 2 + 1] = red1[0] + fc_b[1];
    }
}

// ---------------------------------------------------------------------------
extern "C" void kernel_launch(void* const* d_in, const int* in_sizes, int n_in,
                              void* d_out, int out_size, void* d_ws, size_t ws_size,
                              hipStream_t stream) {
    (void)in_sizes; (void)n_in; (void)out_size; (void)ws_size;

    const float* x      = (const float*)d_in[0];
    const float* enc_W  = (const float*)d_in[1];
    const float* enc_b  = (const float*)d_in[2];
    const float* Wq     = (const float*)d_in[3];
    const float* bq     = (const float*)d_in[4];
    const float* Wk     = (const float*)d_in[5];
    const float* bk     = (const float*)d_in[6];
    const float* Wv     = (const float*)d_in[7];
    const float* bv     = (const float*)d_in[8];
    const float* conv_w = (const float*)d_in[9];
    const float* conv_b = (const float*)d_in[10];
    const float* fc_w   = (const float*)d_in[11];
    const float* fc_b   = (const float*)d_in[12];

    char* ws = (char*)d_ws;
    const size_t Q_OFF   = (size_t)BDIM * CDIM * EDIM * 2;
    const size_t K_OFF   = Q_OFF + (size_t)BDIM * CDIM * ADIM * 2;
    const size_t V_OFF   = K_OFF + (size_t)BDIM * CDIM * ADIM * 2;
    const size_t ATT_OFF = V_OFF + (size_t)BDIM * CDIM * ADIM * 2;

    __bf16* enc = (__bf16*)(ws);
    __bf16* qb  = (__bf16*)(ws + Q_OFF);
    __bf16* kb  = (__bf16*)(ws + K_OFF);
    __bf16* vb  = (__bf16*)(ws + V_OFF);
    float*  atb = (float*)(ws + ATT_OFF);

    enc_kernel<<<dim3(BDIM / 128, EDIM / 128, CDIM), 256, 0, stream>>>(
        x, enc_W, enc_b, enc);

    dim3 qkv_grid(BDIM * CDIM / 128, ADIM / 128);
    qkv_kernel<<<qkv_grid, 256, 0, stream>>>(enc, Wq, bq, qb);
    qkv_kernel<<<qkv_grid, 256, 0, stream>>>(enc, Wk, bk, kb);
    qkv_kernel<<<qkv_grid, 256, 0, stream>>>(enc, Wv, bv, vb);

    const size_t attn_lds =
        (size_t)(64 * QK_STRIDE * 2) * 2 +   // sQ + sK
        (size_t)(64 * S_STRIDE * 4) +        // scores f32
        (size_t)(64 * P_STRIDE * 2) +        // probs bf16
        (size_t)(256 * VT_STRIDE * 2);       // v transposed
    attn_kernel<<<BDIM, 256, attn_lds, stream>>>(qb, kb, vb, atb);

    conv_fc_kernel<<<BDIM, 256, 0, stream>>>(atb, conv_w, conv_b, fc_w, fc_b,
                                             (float*)d_out);
}
